// ContrastiveLoss_zablo_5205500363322
// MI455X (gfx1250) — matrix operand order repaired and checked by reading, not compiled, not run on hardware
//
#include <hip/hip_runtime.h>
#include <hip/hip_bf16.h>

typedef _Float16 v16h __attribute__((ext_vector_type(16)));
typedef _Float16 v8h  __attribute__((ext_vector_type(8)));
typedef _Float16 v4h  __attribute__((ext_vector_type(4)));
typedef float    v8f  __attribute__((ext_vector_type(8)));

#define NROWS 4096          // N (batch)
#define TWO_N 8192          // 2N rows of z
#define DDIM  1024          // embedding dim (K)
#define INV_T 2.0f          // 1 / TEMPERATURE
#define EXP2_SCALE 2.8853900817779268f   // INV_T * log2(e)
#define LN2 0.6931471805599453f
#define LDS_STRIDE 1032     // DDIM + 8 halves pad -> conflict-free 16-lane frag reads
#define ROWS_PER_BLK 128    // 8 waves * 16 rows
#define COL_CHUNK 1024      // columns per block (8 chunks cover 8192)

union FragU { v16h v; v8h h[2]; };

// ---------------- Kernel 1: L2-normalize rows, emit f16 z ----------------
__global__ void ntxent_normalize_f16(const float* __restrict__ emb_i,
                                     const float* __restrict__ emb_j,
                                     _Float16* __restrict__ zh) {
    __shared__ float red[256];
    const int row = blockIdx.x;                 // 0..8191
    const int tid = threadIdx.x;                // 0..255, 4 elems each
    const float* src = (row < NROWS) ? (emb_i + (size_t)row * DDIM)
                                     : (emb_j + (size_t)(row - NROWS) * DDIM);
    const float4 x = *(const float4*)(src + tid * 4);
    float ss = x.x * x.x + x.y * x.y + x.z * x.z + x.w * x.w;
    red[tid] = ss;
    __syncthreads();
    for (int s = 128; s > 0; s >>= 1) {
        if (tid < s) red[tid] += red[tid + s];
        __syncthreads();
    }
    const float scale = 1.0f / fmaxf(sqrtf(red[0]), 1e-12f);
    v4h o;
    o[0] = (_Float16)(x.x * scale);
    o[1] = (_Float16)(x.y * scale);
    o[2] = (_Float16)(x.z * scale);
    o[3] = (_Float16)(x.w * scale);
    *(v4h*)(zh + (size_t)row * DDIM + tid * 4) = o;
}

// ---------------- Kernel 2: zero accumulators ----------------
__global__ void ntxent_zero_accum(float* __restrict__ p, int n) {
    int i = blockIdx.x * blockDim.x + threadIdx.x;
    if (i < n) p[i] = 0.0f;
}

// ---------------- Kernel 3: fused sim GEMM + masked exp-sum + positives ----
// grid: (64 row tiles, 8 column chunks); block: 256 threads = 8 waves.
__global__ void ntxent_sim_wmma(const _Float16* __restrict__ zh,
                                float* __restrict__ denom,
                                float* __restrict__ posArr) {
    extern __shared__ _Float16 ldsA[];          // 128 x 1032 halves = 258 KB

    const int tid  = threadIdx.x;
    const int lane = tid & 31;
    const int wave = tid >> 5;                  // 0..7
    const int rowBlk = blockIdx.x * ROWS_PER_BLK;
    const int colBlk = blockIdx.y * COL_CHUNK;

    // Cooperative stage of the 128 x 1024 A-strip into LDS (16B chunks).
    for (int c = tid; c < ROWS_PER_BLK * (DDIM / 8); c += 256) {
        const int r  = c >> 7;                  // 128 chunks per row
        const int c8 = c & 127;
        *(v8h*)(ldsA + r * LDS_STRIDE + c8 * 8) =
            *(const v8h*)(zh + (size_t)(rowBlk + r) * DDIM + c8 * 8);
    }
    __syncthreads();

    const int rowWave = rowBlk + wave * 16;     // this wave's 16-row tile (mult of 16)
    const int l15 = lane & 15;
    const int hi  = lane >> 4;                  // lane half selector
    // The positive columns for this entire 16-row tile live in exactly one
    // 16-column tile (N=4096 is a multiple of 16):
    const int posTile = (rowWave < NROWS) ? (rowWave + NROWS) : (rowWave - NROWS);
    // Diagonal elements live only in the tile colBase == rowWave.
    const int diagElem = l15 - hi * 8;          // element v on the diagonal for this lane

    float dAcc[8];
    #pragma unroll
    for (int v = 0; v < 8; ++v) dAcc[v] = 0.0f;

    // Loop over 64 column tiles of 16 within this chunk.
    #pragma unroll 1
    for (int ct = 0; ct < COL_CHUNK / 16; ++ct) {
        const int colBase = colBlk + ct * 16;
        const _Float16* bRow = zh + (size_t)(colBase + l15) * DDIM + hi * 8;

        v8f acc = {};
        // K loop: 32 steps of 16x16x32 f16 WMMA.
        #pragma unroll 4
        for (int k0 = 0; k0 < DDIM; k0 += 32) {
            // A fragment (ISA 16-bit A 16x32 layout):
            //   lanes 0-15: K[0..7] then K[16..23]; lanes 16-31: +8.
            const _Float16* ap = ldsA + (wave * 16 + l15) * LDS_STRIDE + k0 + hi * 8;
            FragU a;
            a.h[0] = *(const v8h*)(ap);
            a.h[1] = *(const v8h*)(ap + 16);
            // B fragment (32x16): lane holds column (colBase+l15),
            // K[k0+hi*8..+7] then K[k0+16+hi*8..+7] (same order as A).
            FragU b;
            b.h[0] = *(const v8h*)(bRow + k0);
            b.h[1] = *(const v8h*)(bRow + k0 + 16);
            acc = __builtin_amdgcn_wmma_f32_16x16x32_f16(
                false, a.v, false, b.v, (short)0, acc, false, false);
        }

        // ---- Epilogue (C layout: VGPR v -> M = v + hi*8, lane&15 -> N) ----
        // exp(s/t) == exp2(s * EXP2_SCALE); |arg| <= ~2.9, so the raw
        // hardware v_exp_f32 is safe (no subnormal-range fixup needed).
        if (colBase == rowWave) {
            // Diagonal tile: drop the single self-similarity element per row.
            #pragma unroll
            for (int v = 0; v < 8; ++v) {
                const float e = __builtin_amdgcn_exp2f(acc[v] * EXP2_SCALE);
                dAcc[v] += (v == diagElem) ? 0.0f : e;
            }
        } else {
            #pragma unroll
            for (int v = 0; v < 8; ++v)
                dAcc[v] += __builtin_amdgcn_exp2f(acc[v] * EXP2_SCALE);
        }
        if (colBase == posTile) {
            // Each lane owns at most one positive element: v = l15 - hi*8.
            if ((l15 >> 3) == hi) {
                const int idx = l15 & 7;
                float s = acc[0];
                #pragma unroll
                for (int v = 1; v < 8; ++v) s = (idx == v) ? acc[v] : s;
                posArr[rowWave + l15] = s;   // raw cosine sim; scaled by 1/t later
            }
        }
    }

    // Per-row sums: butterfly within each 16-lane half, then one atomic per row.
    #pragma unroll
    for (int v = 0; v < 8; ++v) {
        float x = dAcc[v];
        x += __shfl_xor(x, 1, 32);
        x += __shfl_xor(x, 2, 32);
        x += __shfl_xor(x, 4, 32);
        x += __shfl_xor(x, 8, 32);
        if (l15 == 0) atomicAdd(&denom[rowWave + v + hi * 8], x);
    }
}

// ---------------- Kernel 4: final scalar reduction ----------------
__global__ void ntxent_final_loss(const float* __restrict__ denom,
                                  const float* __restrict__ posArr,
                                  float* __restrict__ out) {
    __shared__ float red[256];
    const int tid = threadIdx.x;
    float part = 0.0f;
    for (int i = tid; i < TWO_N; i += 256) {
        // ln(denom) = log2(denom) * ln(2); denom ~ 1e4, well inside normal range.
        const float ln_d = __builtin_amdgcn_logf(denom[i]) * LN2;
        part += -(posArr[i] * INV_T - ln_d);
    }
    red[tid] = part;
    __syncthreads();
    for (int s = 128; s > 0; s >>= 1) {
        if (tid < s) red[tid] += red[tid + s];
        __syncthreads();
    }
    if (tid == 0) out[0] = red[0] / (float)TWO_N;
}

extern "C" void kernel_launch(void* const* d_in, const int* in_sizes, int n_in,
                              void* d_out, int out_size, void* d_ws, size_t ws_size,
                              hipStream_t stream) {
    const float* emb_i = (const float*)d_in[0];
    const float* emb_j = (const float*)d_in[1];
    float* out = (float*)d_out;

    // Workspace layout:
    //   [0, 16MB)          : z in f16, 8192 x 1024
    //   [16MB, +32KB)      : denom[8192] f32
    //   [16MB+32KB, +32KB) : pos[8192]   f32
    _Float16* zh   = (_Float16*)d_ws;
    float* denom   = (float*)((char*)d_ws + (size_t)TWO_N * DDIM * sizeof(_Float16));
    float* posArr  = denom + TWO_N;

    ntxent_normalize_f16<<<TWO_N, 256, 0, stream>>>(emb_i, emb_j, zh);
    ntxent_zero_accum<<<(2 * TWO_N + 255) / 256, 256, 0, stream>>>(denom, 2 * TWO_N);

    const size_t ldsBytes = (size_t)ROWS_PER_BLK * LDS_STRIDE * sizeof(_Float16); // 258 KB
    dim3 grid(TWO_N / ROWS_PER_BLK, TWO_N / COL_CHUNK, 1);                        // 64 x 8
    ntxent_sim_wmma<<<grid, 256, ldsBytes, stream>>>(zh, denom, posArr);

    ntxent_final_loss<<<1, 256, 0, stream>>>(denom, posArr, out);
}
